// FocusedLinearAttention_34995393528445
// MI455X (gfx1250) — matrix operand order repaired and checked by reading, not compiled, not run on hardware
//
#include <hip/hip_runtime.h>
#include <hip/hip_bf16.h>
#include <math.h>

// ---------- problem constants ----------
#define B_    16
#define C_    256
#define NH    8
#define HD    32
#define NTOK  4096   // 64*64
#define NBH   (B_*NH)          // 128
typedef unsigned int  u32;
typedef unsigned short u16;

// ---------- WMMA types ----------
typedef __attribute__((ext_vector_type(16))) __bf16 v16bf;
typedef __attribute__((ext_vector_type(8)))  float  v8f;

union Frag { uint4 q[2]; u32 u[8]; v16bf v; };

__device__ __forceinline__ v8f wmma_bf16(const Frag& a, const Frag& b, v8f c) {
    return __builtin_amdgcn_wmma_f32_16x16x32_bf16(false, a.v, false, b.v,
                                                   (short)0, c, false, false);
}

// fp32 -> bf16 (RTNE) via bit math
__device__ __forceinline__ u16 f2bf(float f) {
    u32 u; __builtin_memcpy(&u, &f, 4);
    u32 r = (u + 0x7FFFu + ((u >> 16) & 1u)) >> 16;
    return (u16)r;
}
__device__ __forceinline__ float bf2f(u16 h) {
    u32 u = ((u32)h) << 16; float f; __builtin_memcpy(&f, &u, 4);
    return f;
}

// ============================================================
// P0: pack Wq (256x256) | Wkv (512x256) | Wproj (256x256) -> bf16
//     wbf rows: [0,256)=Wq, [256,768)=Wkv, [768,1024)=Wproj
// ============================================================
__global__ void k_pack_weights(const float* __restrict__ Wq, const float* __restrict__ Wkv,
                               const float* __restrict__ Wp, u16* __restrict__ wbf) {
    int idx = blockIdx.x * blockDim.x + threadIdx.x;   // 1024*256
    int r = idx >> 8, c = idx & 255;
    float v;
    if (r < 256)       v = Wq[(size_t)r * C_ + c];
    else if (r < 768)  v = Wkv[(size_t)(r - 256) * C_ + c];
    else               v = Wp[(size_t)(r - 768) * C_ + c];
    wbf[idx] = f2bf(v);
}

// P1: rs[c] = 1 / softplus(scale_param[c])
__global__ void k_softplus_inv(const float* __restrict__ sp, float* __restrict__ rs) {
    int c = threadIdx.x;
    rs[c] = 1.0f / log1pf(expf(sp[c]));
}

// ============================================================
// K0: x (B,C,N) fp32 -> xt (B,N,C) bf16   (LDS tiled transpose)
// grid (N/32, C/32, B), block (32,8)
// ============================================================
__global__ void k_transpose_cast(const float* __restrict__ x, u16* __restrict__ xbf) {
    __shared__ float tile[32][33];
    int b  = blockIdx.z;
    int c0 = blockIdx.y * 32;
    int n0 = blockIdx.x * 32;
    int tx = threadIdx.x, ty = threadIdx.y;
    const float* xp = x + ((size_t)b * C_ + c0) * NTOK + n0;
#pragma unroll
    for (int i = 0; i < 32; i += 8)
        tile[ty + i][tx] = xp[(size_t)(ty + i) * NTOK + tx];
    __syncthreads();
    u16* op = xbf + ((size_t)b * NTOK + n0) * C_ + c0;
#pragma unroll
    for (int i = 0; i < 32; i += 8) {
        int nl = ty + i;
        op[(size_t)nl * C_ + tx] = f2bf(tile[tx][nl]);
    }
}

// ============================================================
// K1: fused QKV GEMM. one wave computes 16x64 (4 col tiles).
//     A = xt bf16, B = packed weights bf16. cols [0,256)=q,[256,512)=k,[512,768)=v
//     4096 token tiles x 12 col groups = 49152 waves
// ============================================================
__global__ void k_qkv_gemm(const u16* __restrict__ xbf, const u16* __restrict__ wbf,
                           const float* __restrict__ bq, const float* __restrict__ bkv,
                           float* __restrict__ qf, float* __restrict__ kf, float* __restrict__ vf) {
    int wave = (blockIdx.x * blockDim.x + threadIdx.x) >> 5;
    int lane = threadIdx.x & 31;
    int tm = wave / 12;                  // token tile over B*N
    int g  = wave % 12;                  // 64-col group
    int cb = g * 64;
    int m    = lane & 15;
    int half = lane >> 4;
    int koffA = half ? 8 : 0;            // A fragment dword offset
    int boff  = half ? 8 : 0;            // B fragment dword offset
    const u32* ap = (const u32*)xbf + (((size_t)(tm * 16 + m)) * C_ >> 1);
    const u32* bp[4];
    float biasv[4];
#pragma unroll
    for (int t = 0; t < 4; t++) {
        int colt = cb + t * 16 + m;
        bp[t] = (const u32*)wbf + ((size_t)colt * C_ >> 1);
        biasv[t] = (g < 4) ? bq[colt] : bkv[colt - C_];
    }
    v8f acc[4] = {};
    for (int ks = 0; ks < C_; ks += 32) {
        int kd = ks >> 1;
        Frag A;
        A.q[0] = ((const uint4*)(ap + kd + koffA))[0];
        A.q[1] = ((const uint4*)(ap + kd + koffA + 4))[0];
#pragma unroll
        for (int t = 0; t < 4; t++) {
            Frag Bf;
            Bf.q[0] = ((const uint4*)(bp[t] + kd + boff))[0];
            Bf.q[1] = ((const uint4*)(bp[t] + kd + boff + 4))[0];
            acc[t] = wmma_bf16(A, Bf, acc[t]);
        }
    }
    float* dst; int cb2;                 // wave-uniform select
    if (g < 4)       { dst = qf; cb2 = cb; }
    else if (g < 8)  { dst = kf; cb2 = cb - 256; }
    else             { dst = vf; cb2 = cb - 512; }
    int ncol = lane & 15;
#pragma unroll
    for (int t = 0; t < 4; t++) {
#pragma unroll
        for (int r = 0; r < 8; r++) {
            int row = tm * 16 + r + (half ? 8 : 0);
            dst[(size_t)row * C_ + cb2 + t * 16 + ncol] = acc[t][r] + biasv[t];
        }
    }
}

// ============================================================
// K2: focusing. one wave per token row (256 ch -> 8 per lane).
// ============================================================
__global__ void k_focus(const float* __restrict__ qf, const float* __restrict__ kf,
                        const float* __restrict__ vf, const float* __restrict__ rs,
                        u16* __restrict__ qbf, u16* __restrict__ kT, u16* __restrict__ vT) {
    int wave = (blockIdx.x * blockDim.x + threadIdx.x) >> 5;   // 0..65535 rows
    int lane = threadIdx.x & 31;
    int b = wave >> 12;
    int n = wave & 4095;
    size_t rb = (size_t)wave * C_;

    float vv[8], q3[8], k3[8];
    float qs = 0.f, ks = 0.f, q3s = 0.f, k3s = 0.f;
#pragma unroll
    for (int j = 0; j < 8; j++) {
        int c = lane + 32 * j;
        float s = rs[c];
        float q = (fmaxf(qf[rb + c], 0.f) + 1e-6f) * s;
        float k = (fmaxf(kf[rb + c], 0.f) + 1e-6f) * s;
        vv[j] = vf[rb + c];
        qs += q * q;  ks += k * k;
        q3[j] = q * q * q;  k3[j] = k * k * k;
        q3s += q3[j] * q3[j];  k3s += k3[j] * k3[j];
    }
#pragma unroll
    for (int off = 16; off; off >>= 1) {
        qs  += __shfl_xor(qs,  off, 32);
        ks  += __shfl_xor(ks,  off, 32);
        q3s += __shfl_xor(q3s, off, 32);
        k3s += __shfl_xor(k3s, off, 32);
    }
    float qsc = sqrtf(qs) * rsqrtf(fmaxf(q3s, 1e-30f));  // ||q|| / ||q^3||
    float ksc = sqrtf(ks) * rsqrtf(fmaxf(k3s, 1e-30f));
#pragma unroll
    for (int j = 0; j < 8; j++) {
        int c = lane + 32 * j;
        int h = c >> 5, d = c & 31;
        qbf[rb + c] = f2bf(q3[j] * qsc);
        size_t tix = ((size_t)((b * NH + h) * HD + d)) * NTOK + n;
        kT[tix] = f2bf(k3[j] * ksc);
        vT[tix] = f2bf(vv[j]);
    }
}

// ============================================================
// K3: ksum[bh][d] = sum_n kT[bh][d][n].  one block per bh.
// ============================================================
__global__ void k_ksum(const u16* __restrict__ kT, float* __restrict__ ksum) {
    int bh = blockIdx.x;
    int t  = threadIdx.x;
    int d  = t >> 3;
    int j  = t & 7;
    const u16* kp = kT + ((size_t)bh * HD + d) * NTOK;
    float s = 0.f;
    for (int i = 0; i < 512; i++) s += bf2f(kp[j + 8 * i]);
    __shared__ float red[256];
    red[t] = s;
    __syncthreads();
    if (t < 32) {
        float tot = 0.f;
#pragma unroll
        for (int jj = 0; jj < 8; jj++) tot += red[t * 8 + jj];
        ksum[bh * HD + t] = tot;
    }
}

// ============================================================
// K4: kv_state = k^T v per head (32x32, K=4096). 8 waves/bh split K,
//     LDS reduce, store transposed bf16 kvT[bh][d][c].
// ============================================================
__global__ void k_kvstate(const u16* __restrict__ kT, const u16* __restrict__ vT,
                          u16* __restrict__ kvT) {
    int bh   = blockIdx.x;
    int t    = threadIdx.x;
    int lane = t & 31;
    int w    = t >> 5;                 // wave 0..7 -> K slice
    int m    = lane & 15;
    int half = lane >> 4;
    int koffA = half ? 8 : 0;
    int boff  = half ? 8 : 0;
    const u32* a0 = (const u32*)kT + ((((size_t)bh * HD +      m) * NTOK) >> 1);
    const u32* a1 = (const u32*)kT + ((((size_t)bh * HD + 16 + m) * NTOK) >> 1);
    const u32* b0 = (const u32*)vT + ((((size_t)bh * HD +      m) * NTOK) >> 1);
    const u32* b1 = (const u32*)vT + ((((size_t)bh * HD + 16 + m) * NTOK) >> 1);
    v8f acc[4] = {};                    // [A0B0, A0B1, A1B0, A1B1]
    int ks0 = w * 512;
    for (int ks = ks0; ks < ks0 + 512; ks += 32) {
        int kd = ks >> 1;
        Frag A0, A1, B0, B1;
        A0.q[0] = ((const uint4*)(a0 + kd + koffA))[0];
        A0.q[1] = ((const uint4*)(a0 + kd + koffA + 4))[0];
        A1.q[0] = ((const uint4*)(a1 + kd + koffA))[0];
        A1.q[1] = ((const uint4*)(a1 + kd + koffA + 4))[0];
        B0.q[0] = ((const uint4*)(b0 + kd + boff))[0];
        B0.q[1] = ((const uint4*)(b0 + kd + boff + 4))[0];
        B1.q[0] = ((const uint4*)(b1 + kd + boff))[0];
        B1.q[1] = ((const uint4*)(b1 + kd + boff + 4))[0];
        acc[0] = wmma_bf16(A0, B0, acc[0]);
        acc[1] = wmma_bf16(A0, B1, acc[1]);
        acc[2] = wmma_bf16(A1, B0, acc[2]);
        acc[3] = wmma_bf16(A1, B1, acc[3]);
    }
    __shared__ float red[8][4][8][32];   // 32 KB
#pragma unroll
    for (int tile = 0; tile < 4; tile++)
#pragma unroll
        for (int r = 0; r < 8; r++)
            red[w][tile][r][lane] = acc[tile][r];
    __syncthreads();
    if (w == 0) {
        u16* outp = kvT + (size_t)bh * HD * HD;
        int ncol = lane & 15;
#pragma unroll
        for (int tile = 0; tile < 4; tile++) {
#pragma unroll
            for (int r = 0; r < 8; r++) {
                float s = 0.f;
#pragma unroll
                for (int ww = 0; ww < 8; ww++) s += red[ww][tile][r][lane];
                int mr = r + (half ? 8 : 0);
                int c = mr   + ((tile & 2) ? 16 : 0);
                int d = ncol + ((tile & 1) ? 16 : 0);
                outp[d * HD + c] = f2bf(s);     // transposed store
            }
        }
    }
}

// ============================================================
// K5: z[bh][n] = 1/(q . ksum + 1e-6)
// ============================================================
__global__ void k_z(const u16* __restrict__ qbf, const float* __restrict__ ksum,
                    float* __restrict__ zbuf) {
    int idx = blockIdx.x * blockDim.x + threadIdx.x;   // 128*4096
    int bh = idx >> 12, n = idx & 4095;
    int b = bh >> 3, h = bh & 7;
    const u16*  qp = qbf + ((size_t)(b * NTOK + n)) * C_ + h * HD;
    const float* kp = ksum + bh * HD;
    float s = 0.f;
#pragma unroll
    for (int d = 0; d < HD; d++) s += bf2f(qp[d]) * kp[d];
    zbuf[idx] = 1.0f / (s + 1e-6f);
}

// ============================================================
// K6: out = (q @ kv_state) * z  -> outpre fp32 [B][N][C]
// ============================================================
__global__ void k_out(const u16* __restrict__ qbf, const u16* __restrict__ kvT,
                      const float* __restrict__ zbuf, float* __restrict__ outpre) {
    int wave = (blockIdx.x * blockDim.x + threadIdx.x) >> 5;   // 128*256 waves
    int lane = threadIdx.x & 31;
    int bh = wave >> 8;
    int tt = wave & 255;
    int b = bh >> 3, h = bh & 7;
    int m = lane & 15, half = lane >> 4;
    int koffA = half ? 8 : 0;
    int boff  = half ? 8 : 0;
    int tok0 = tt * 16;
    const u32* ap  = (const u32*)qbf + ((((size_t)(b * NTOK + tok0 + m)) * C_ + h * HD) >> 1);
    const u32* bp0 = (const u32*)kvT + ((((size_t)bh * HD * HD) + (size_t)m * HD) >> 1);
    const u32* bp1 = (const u32*)kvT + ((((size_t)bh * HD * HD) + (size_t)(16 + m) * HD) >> 1);
    Frag A, B0, B1;
    A.q[0]  = ((const uint4*)(ap + koffA))[0];
    A.q[1]  = ((const uint4*)(ap + koffA + 4))[0];
    B0.q[0] = ((const uint4*)(bp0 + boff))[0];
    B0.q[1] = ((const uint4*)(bp0 + boff + 4))[0];
    B1.q[0] = ((const uint4*)(bp1 + boff))[0];
    B1.q[1] = ((const uint4*)(bp1 + boff + 4))[0];
    v8f acc0 = {}, acc1 = {};
    acc0 = wmma_bf16(A, B0, acc0);
    acc1 = wmma_bf16(A, B1, acc1);
    int ncol = lane & 15;
#pragma unroll
    for (int r = 0; r < 8; r++) {
        int tok = tok0 + r + (half ? 8 : 0);
        float z = zbuf[bh * NTOK + tok];
        size_t orow = ((size_t)(b * NTOK + tok)) * C_ + h * HD;
        outpre[orow + ncol]      = acc0[r] * z;
        outpre[orow + 16 + ncol] = acc1[r] * z;
    }
}

// ============================================================
// K7: depthwise 5x5 conv over vT bf16; outbf = bf16(outpre + conv + bias)
// ============================================================
__global__ void k_dwconv(const u16* __restrict__ vT, const float* __restrict__ w,
                         const float* __restrict__ bias, const float* __restrict__ outpre,
                         u16* __restrict__ outbf) {
    int idx = blockIdx.x * blockDim.x + threadIdx.x;   // 2^24
    int n  = idx & 4095;
    int dd = (idx >> 12) & 31;
    int bh = idx >> 17;
    int y = n >> 6, x2 = n & 63;
    const u16*  vp = vT + ((size_t)bh * HD + dd) * NTOK;
    const float* wp = w + dd * 25;
    float s = bias[dd];
#pragma unroll
    for (int ky = 0; ky < 5; ky++) {
        int yy = y + ky - 2;
        if (yy < 0 || yy >= 64) continue;
#pragma unroll
        for (int kx = 0; kx < 5; kx++) {
            int xx = x2 + kx - 2;
            if (xx < 0 || xx >= 64) continue;
            s += bf2f(vp[yy * 64 + xx]) * wp[ky * 5 + kx];
        }
    }
    int b = bh >> 3, h = bh & 7;
    size_t o = ((size_t)(b * NTOK + n)) * C_ + h * HD + dd;
    outbf[o] = f2bf(outpre[o] + s);
}

// ============================================================
// K8: proj GEMM (both operands bf16) + bias + transposed store (B,C,H,W)
//     one wave computes 16x64 (4 col tiles); 4096 x 4 = 16384 waves
// ============================================================
__global__ void k_proj(const u16* __restrict__ outbf, const u16* __restrict__ wbf,
                       const float* __restrict__ bp, float* __restrict__ out) {
    int wave = (blockIdx.x * blockDim.x + threadIdx.x) >> 5;
    int lane = threadIdx.x & 31;
    int tm = wave >> 2;                  // token tile over B*N
    int g  = wave & 3;                   // 64-col group
    int cb = g * 64;
    int m = lane & 15, half = lane >> 4;
    int koffA = half ? 8 : 0;
    int boff  = half ? 8 : 0;
    const u32* ap = (const u32*)outbf + (((size_t)(tm * 16 + m)) * C_ >> 1);
    const u32* bpr[4];
    float biasv[4];
#pragma unroll
    for (int t = 0; t < 4; t++) {
        int colt = cb + t * 16 + m;
        bpr[t] = (const u32*)wbf + (((size_t)(768 + colt)) * C_ >> 1);  // Wproj rows
        biasv[t] = bp[colt];
    }
    v8f acc[4] = {};
    for (int ks = 0; ks < C_; ks += 32) {
        int kd = ks >> 1;
        Frag A;
        A.q[0] = ((const uint4*)(ap + kd + koffA))[0];
        A.q[1] = ((const uint4*)(ap + kd + koffA + 4))[0];
#pragma unroll
        for (int t = 0; t < 4; t++) {
            Frag Bf;
            Bf.q[0] = ((const uint4*)(bpr[t] + kd + boff))[0];
            Bf.q[1] = ((const uint4*)(bpr[t] + kd + boff + 4))[0];
            acc[t] = wmma_bf16(A, Bf, acc[t]);
        }
    }
    int ncol = lane & 15;
#pragma unroll
    for (int t = 0; t < 4; t++) {
#pragma unroll
        for (int r = 0; r < 8; r++) {
            int grow = tm * 16 + r + (half ? 8 : 0);
            int n  = grow & 4095;
            int bb = grow >> 12;
            int co = cb + t * 16 + ncol;
            out[((size_t)bb * C_ + co) * NTOK + n] = acc[t][r] + biasv[t];
        }
    }
}

// ============================================================
extern "C" void kernel_launch(void* const* d_in, const int* in_sizes, int n_in,
                              void* d_out, int out_size, void* d_ws, size_t ws_size,
                              hipStream_t stream) {
    (void)in_sizes; (void)n_in; (void)out_size; (void)ws_size;
    const float* x      = (const float*)d_in[0];
    const float* Wq     = (const float*)d_in[1];
    const float* bq     = (const float*)d_in[2];
    const float* Wkv    = (const float*)d_in[3];
    const float* bkv    = (const float*)d_in[4];
    const float* Wproj  = (const float*)d_in[5];
    const float* bproj  = (const float*)d_in[6];
    const float* dwc_w  = (const float*)d_in[7];
    const float* dwc_b  = (const float*)d_in[8];
    const float* scalep = (const float*)d_in[9];
    float* out = (float*)d_out;

    const size_t nE = (size_t)B_ * NTOK * C_;          // 16,777,216 elems
    char* ws = (char*)d_ws;
    size_t off = 0;
    u16*  xbf    = (u16*)(ws + off);  off += nE * 2;
    u16*  wbf    = (u16*)(ws + off);  off += (size_t)1024 * C_ * 2;
    float* rs    = (float*)(ws + off); off += C_ * 4;
    off = (off + 255) & ~(size_t)255;
    float* qf    = (float*)(ws + off); off += nE * 4;
    float* kf    = (float*)(ws + off); off += nE * 4;
    float* vf    = (float*)(ws + off); off += nE * 4;
    u16*  qbf    = (u16*)(ws + off);  off += nE * 2;
    u16*  kT     = (u16*)(ws + off);  off += nE * 2;
    u16*  vT     = (u16*)(ws + off);  off += nE * 2;
    float* ksum  = (float*)(ws + off); off += (size_t)NBH * HD * 4;
    u16*  kvT    = (u16*)(ws + off);  off += (size_t)NBH * HD * HD * 2;
    off = (off + 255) & ~(size_t)255;
    float* zbuf  = (float*)(ws + off); off += (size_t)NBH * NTOK * 4;
    float* outpre= (float*)(ws + off); off += nE * 4;
    u16*  outbf  = (u16*)(ws + off);  off += nE * 2;

    k_pack_weights<<<1024, 256, 0, stream>>>(Wq, Wkv, Wproj, wbf);
    k_softplus_inv<<<1, 256, 0, stream>>>(scalep, rs);
    k_transpose_cast<<<dim3(NTOK / 32, C_ / 32, B_), dim3(32, 8), 0, stream>>>(x, xbf);
    // 49152 waves (4096 token tiles x 12 col groups)
    k_qkv_gemm<<<6144, 256, 0, stream>>>(xbf, wbf, bq, bkv, qf, kf, vf);
    k_focus<<<8192, 256, 0, stream>>>(qf, kf, vf, rs, qbf, kT, vT);
    k_ksum<<<NBH, 256, 0, stream>>>(kT, ksum);
    k_kvstate<<<NBH, 256, 0, stream>>>(kT, vT, kvT);
    k_z<<<(NBH * NTOK) / 256, 256, 0, stream>>>(qbf, ksum, zbuf);
    k_out<<<4096, 256, 0, stream>>>(qbf, kvT, zbuf, outpre);
    k_dwconv<<<(NBH * HD * NTOK) / 256, 256, 0, stream>>>(vT, dwc_w, dwc_b, outpre, outbf);
    // 16384 waves (4096 token tiles x 4 col groups)
    k_proj<<<2048, 256, 0, stream>>>(outbf, wbf, bproj, out);
}
